// NeuralSparse_44753559224382
// MI455X (gfx1250) — matrix operand order repaired and checked by reading, not compiled, not run on hardware
//
#include <hip/hip_runtime.h>

#define KSAMP 5

typedef __attribute__((ext_vector_type(16))) __bf16 v16bf;
typedef __attribute__((ext_vector_type(8)))  float  v8f;

union BFrag {
    v16bf    v;
    unsigned u[8];
    uint4    q[2];
};

__device__ __forceinline__ unsigned short f2bf(float f) {
    unsigned b = __float_as_uint(f);
    return (unsigned short)((b + 0x7FFFu + ((b >> 16) & 1u)) >> 16);  // RNE
}

// Deterministic counter-hash gumbel; tau = 1 so z = logit + g.
__device__ __forceinline__ float gumbel_z(float logit, unsigned e, unsigned k) {
    unsigned s = e * 0x9E3779B9u ^ (k * 0x85EBCA6Bu + 0x6C62272Eu);
    s ^= s >> 16; s *= 0x7FEB352Du;
    s ^= s >> 15; s *= 0x846CA68Bu;
    s ^= s >> 16;
    float u = ((float)(s >> 8) + 0.5f) * (1.0f / 16777216.0f);  // (0,1)
    float g = -__logf(-__logf(u));
    return logit + g;
}

// Order-preserving float -> uint map (for atomicMax on floats).
__device__ __forceinline__ unsigned ford(float f) {
    unsigned b = __float_as_uint(f);
    return (b & 0x80000000u) ? ~b : (b | 0x80000000u);
}

__device__ __forceinline__ int edge_src(const int* ei, const int* ec, int E_intra, int e) {
    return (e < E_intra) ? ei[2 * e] : ec[2 * (e - E_intra)];
}

// VALU-only 16-lane row reduction: v_add_f32_dpp row_ror:{1,2,4,8}.
template <int CTRL>
__device__ __forceinline__ float dpp_add(float x) {
    int y = __builtin_amdgcn_update_dpp(0, __float_as_int(x), CTRL, 0xF, 0xF, false);
    return x + __int_as_float(y);
}
__device__ __forceinline__ float row16_sum(float p) {
    p = dpp_add<0x121>(p);  // row_ror:1
    p = dpp_add<0x122>(p);  // row_ror:2
    p = dpp_add<0x124>(p);  // row_ror:4
    p = dpp_add<0x128>(p);  // row_ror:8
    return p;
}

// ---- CDNA5 async global->LDS gather (ASYNCcnt-tracked) ----
__device__ __forceinline__ void async_load16(unsigned ldsOff, unsigned long long gaddr) {
    asm volatile("global_load_async_to_lds_b128 %0, %1, off"
                 :: "v"(ldsOff), "v"(gaddr)
                 : "memory");
}

__device__ __forceinline__ void wait_async_le2() {
#if __has_builtin(__builtin_amdgcn_s_wait_asynccnt)
    __builtin_amdgcn_s_wait_asynccnt(2);
#else
    asm volatile("s_wait_asynccnt 0x2" ::: "memory");
#endif
}

// ---------------- setup kernels ----------------

__global__ void k_prep_x(const float* __restrict__ x, unsigned short* __restrict__ xbf, int n) {
    int i = blockIdx.x * blockDim.x + threadIdx.x;
    if (i < n) xbf[i] = f2bf(x[i]);
}

// Pack W1 (256x128 f32, K-major) into per-lane bf16 B fragments:
// Bp[((nt*8+kk)*32+lane)*8 + v] = pack(W1[k][c], W1[k+1][c])
//   n = lane%16, khalf = lane/16, k = kk*32 + khalf*16 + 2v, c = nt*16 + n
__global__ void k_pack_w1(const float* __restrict__ W1, unsigned* __restrict__ Bp) {
    int i = blockIdx.x * blockDim.x + threadIdx.x;  // 16384 total
    if (i >= 8 * 8 * 32 * 8) return;
    int v    = i & 7;
    int lane = (i >> 3) & 31;
    int kk   = (i >> 8) & 7;
    int nt   = i >> 11;
    int n     = lane & 15;
    int khalf = lane >> 4;
    int k = kk * 32 + khalf * 16 + 2 * v;
    int c = nt * 16 + n;
    unsigned lo = f2bf(W1[k * 128 + c]);
    unsigned hi = f2bf(W1[(k + 1) * 128 + c]);
    Bp[i] = lo | (hi << 16);
}

// ---------------- main GEMM + logits kernel ----------------

__global__ void __launch_bounds__(256) k_gemm_logits(
    const unsigned short* __restrict__ xbf,
    const unsigned*       __restrict__ Bp,
    const int* __restrict__ ei, const int* __restrict__ ec,
    int E_intra, int E, int numTiles,
    const float* __restrict__ b1, const float* __restrict__ W2,
    const float* __restrict__ b2, float* __restrict__ logits) {

    __shared__ unsigned short As[2][16 * 256];  // double-buffered A tile (2 x 8KB)
    __shared__ float llog[16];

    const int tid  = threadIdx.x;
    const int lane = tid & 31;
    const int wave = tid >> 5;      // 8 waves; wave == N-tile (16 output cols each)

    const int r      = tid >> 4;    // edge row within tile (gather role)
    const int cchunk = tid & 15;    // 16B chunk within row (gather role)
    const int m      = lane & 15;   // A-fragment row / C column-in-tile
    const int khalf  = lane >> 4;
    const int col    = (wave << 4) + m;  // global output column 0..127

    const float b1c = b1[col];
    const float w2c = W2[col];
    const float b2v = b2[0];

    // B fragments are tile-invariant: all 8 K-chunks live in registers.
    BFrag bf[8];
#pragma unroll
    for (int kk = 0; kk < 8; ++kk) {
        const uint4* p = (const uint4*)(Bp + ((((unsigned)wave * 8 + kk) * 32 + lane) << 3));
        bf[kk].q[0] = p[0];
        bf[kk].q[1] = p[1];
    }

    if (tid < 16) llog[tid] = 0.0f;

    // LDS byte offsets for this thread's two 16B gather destinations.
    const unsigned ldsRow = ((unsigned)r << 9) + ((unsigned)cchunk << 4);
    const unsigned ldsBuf[2] = { (unsigned)(size_t)&As[0][0] + ldsRow,
                                 (unsigned)(size_t)&As[1][0] + ldsRow };
    const unsigned long long xbase =
        (unsigned long long)(size_t)xbf + ((unsigned long long)(unsigned)cchunk << 4);

    auto issue_tile = [&](int tt, unsigned ldsDst) {
        int e  = (tt << 4) + r;
        int ee = (e < E) ? e : 0;  // clamp tail: harmless gather, never stored
        const int* ep = (ee < E_intra) ? ei : ec;
        int bb  = (ee < E_intra) ? (ee << 1) : ((ee - E_intra) << 1);
        int src = ep[bb];
        int dst = ep[bb + 1];
        async_load16(ldsDst,        xbase + ((unsigned long long)(unsigned)src << 8));
        async_load16(ldsDst + 256u, xbase + ((unsigned long long)(unsigned)dst << 8));
    };

    const unsigned abase0 = (unsigned)m * 256u + ((unsigned)khalf << 4);
    auto load_frag = [&](BFrag& af, int cur, int kk) {
        unsigned abase = abase0 + (unsigned)kk * 32u;
#pragma unroll
        for (int v = 0; v < 4; ++v)
            af.u[v] = *(const unsigned*)&As[cur][abase + 2u * v];
#pragma unroll
        for (int v = 4; v < 8; ++v)
            af.u[v] = *(const unsigned*)&As[cur][abase + 8u + 2u * (v - 4)];
    };

    int cur = 0;
    issue_tile((int)blockIdx.x, ldsBuf[0]);  // prologue: fill buffer 0

    for (int t = (int)blockIdx.x; t < numTiles; t += (int)gridDim.x) {
        // Prefetch next tile into the other buffer, then drain current one.
        issue_tile(t + (int)gridDim.x, ldsBuf[cur ^ 1]);
        wait_async_le2();        // ASYNCcnt <= 2: current buffer's 2 loads landed
        __syncthreads();         // all waves' async writes visible; llog re-zeroed

        // ---- 8x v_wmma_f32_16x16x32_bf16 over K=256, A-frag double-buffered
        //      so fragment kk+1's ds_load_b128 pair is in flight during wmma kk.
        v8f acc = {0.f, 0.f, 0.f, 0.f, 0.f, 0.f, 0.f, 0.f};
        BFrag af[2];
        load_frag(af[0], cur, 0);
#pragma unroll
        for (int kk = 0; kk < 8; ++kk) {
            if (kk + 1 < 8) load_frag(af[(kk + 1) & 1], cur, kk + 1);
            acc = __builtin_amdgcn_wmma_f32_16x16x32_bf16(
                false, af[kk & 1].v, false, bf[kk].v, (short)0, acc, false, false);
        }

        // ---- epilogue: bias + relu + dot(W2) via DPP row reduction ----
#pragma unroll
        for (int v = 0; v < 8; ++v) {
            float h = acc[v] + b1c;
            h = h > 0.0f ? h : 0.0f;
            float p = row16_sum(h * w2c);             // VALU-only reduction
            if (lane == 0)       atomicAdd(&llog[v], p);       // row M=v
            else if (lane == 16) atomicAdd(&llog[v + 8], p);   // row M=v+8
        }
        __syncthreads();

        if (tid < 16) {
            int eo = (t << 4) + tid;
            if (eo < E) logits[eo] = llog[tid] + b2v;
            llog[tid] = 0.0f;    // re-zero by same thread; next barrier orders it
        }
        cur ^= 1;
    }
}

// ---------------- segment gumbel-argmax kernels ----------------

__global__ void k_seg_init(unsigned* __restrict__ segmax, int n) {
    int i = blockIdx.x * blockDim.x + threadIdx.x;
    if (i < n) segmax[i] = 0u;  // ordered-uint minimum
}

__global__ void k_zmax(const float* __restrict__ logits,
                       const int* __restrict__ ei, const int* __restrict__ ec,
                       int E_intra, int E, int N, unsigned* __restrict__ segmax) {
    int e = blockIdx.x * blockDim.x + threadIdx.x;
    if (e >= E) return;
    int src = edge_src(ei, ec, E_intra, e);
    float lg = logits[e];
#pragma unroll
    for (int k = 0; k < KSAMP; ++k) {
        unsigned o = ford(gumbel_z(lg, (unsigned)e, (unsigned)k));
        atomicMax(&segmax[k * N + src], o);
    }
}

__global__ void k_weights(const float* __restrict__ logits,
                          const int* __restrict__ ei, const int* __restrict__ ec,
                          int E_intra, int E, int N,
                          const unsigned* __restrict__ segmax,
                          float* __restrict__ out) {
    int e = blockIdx.x * blockDim.x + threadIdx.x;
    if (e >= E) return;
    int src = edge_src(ei, ec, E_intra, e);
    float lg = logits[e];
    float w = 0.0f;
#pragma unroll
    for (int k = 0; k < KSAMP; ++k) {
        unsigned o = ford(gumbel_z(lg, (unsigned)e, (unsigned)k));
        w += (o == segmax[k * N + src]) ? 1.0f : 0.0f;  // forward of hard ST sample
    }
    out[e] = w;
}

// ---------------- host launcher ----------------

extern "C" void kernel_launch(void* const* d_in, const int* in_sizes, int n_in,
                              void* d_out, int out_size, void* d_ws, size_t ws_size,
                              hipStream_t stream) {
    const float* x  = (const float*)d_in[0];
    const int*   ei = (const int*)d_in[1];
    const int*   ec = (const int*)d_in[2];
    // d_in[3] node_layers: unused by the forward output
    const float* W1 = (const float*)d_in[4];
    const float* b1 = (const float*)d_in[5];
    const float* W2 = (const float*)d_in[6];
    const float* b2 = (const float*)d_in[7];

    const int N       = in_sizes[0] / 128;
    const int E_intra = in_sizes[1] / 2;
    const int E_cross = in_sizes[2] / 2;
    const int E       = E_intra + E_cross;
    const int numTiles = (E + 15) / 16;

    auto al = [](size_t s) { return (s + 255) & ~(size_t)255; };
    char* w = (char*)d_ws;
    unsigned short* xbf = (unsigned short*)w;
    size_t off = al((size_t)N * 128 * 2);
    unsigned* Bp = (unsigned*)(w + off);     off += al((size_t)16384 * 4);
    float* logits = (float*)(w + off);       off += al((size_t)E * 4);
    unsigned* segmax = (unsigned*)(w + off); // K*N uints

    const int nx = N * 128;
    k_prep_x<<<(nx + 255) / 256, 256, 0, stream>>>(x, xbf, nx);
    k_pack_w1<<<16384 / 256, 256, 0, stream>>>(W1, Bp);
    k_gemm_logits<<<2048, 256, 0, stream>>>(xbf, Bp, ei, ec, E_intra, E, numTiles,
                                            b1, W2, b2, logits);
    const int ns = KSAMP * N;
    k_seg_init<<<(ns + 255) / 256, 256, 0, stream>>>(segmax, ns);
    k_zmax<<<(E + 255) / 256, 256, 0, stream>>>(logits, ei, ec, E_intra, E, N, segmax);
    k_weights<<<(E + 255) / 256, 256, 0, stream>>>(logits, ei, ec, E_intra, E, N, segmax,
                                                   (float*)d_out);
}
